// EagerAttention_59184649339001
// MI455X (gfx1250) — compile-verified, matrix operand order checked by reading, not run on hardware
//
#include <hip/hip_runtime.h>

// ---------------------------------------------------------------------------
// EagerAttention for gfx1250 (MI455X): bf16 WMMA pipeline.
// x/w fp32 -> bf16 staging -> WMMA GEMMs (f32 accum) -> RoPE -> flash attn
// (online softmax, causal) -> WMMA output projection -> fp32 out.
// Mask input (d_in[1]) is ignored; causal mask computed analytically.
// GEMM wave tile is 32x64 (8 WMMA per 6 fragment loads) for WMMA density.
// ---------------------------------------------------------------------------

#define TSEQ 2048
#define DMODEL 2048
#define NHEADS 32
#define NKV 8
#define HDIM 64
#define KVDIM (NKV * HDIM)   // 512

typedef __attribute__((ext_vector_type(16))) __bf16 v16bf;
typedef __attribute__((ext_vector_type(8)))  __bf16 v8bf;
typedef __attribute__((ext_vector_type(8)))  float  v8f;

static __device__ __forceinline__ __bf16 to_bf16(float f) {
  unsigned u = __builtin_bit_cast(unsigned, f);
  unsigned r = u + 0x7FFFu + ((u >> 16) & 1u);   // round-to-nearest-even
  unsigned short h = (unsigned short)(r >> 16);
  return __builtin_bit_cast(__bf16, h);
}

// Load one 16x32 bf16 WMMA operand fragment for this lane.
// Per ISA 16-bit A layout (wave32): lane L<16 -> row M=L, K {0..7, 16..23};
// lane L>=16 -> row M=L-16, K {8..15, 24..31}. Two contiguous 16B loads.
// Caller passes p = base + (lane&15)*stride + (lane>>4)*8.
static __device__ __forceinline__ v16bf ldfrag(const __bf16* p) {
  v8bf lo = *(const v8bf*)(p);
  v8bf hi = *(const v8bf*)(p + 16);
  return __builtin_shufflevector(lo, hi, 0, 1, 2, 3, 4, 5, 6, 7,
                                 8, 9, 10, 11, 12, 13, 14, 15);
}

static __device__ __forceinline__ v8f wmma_bf16(v16bf a, v16bf b, v8f c) {
  return __builtin_amdgcn_wmma_f32_16x16x32_bf16(
      false, a, false, b, (short)0, c, false, false);
}

// ---------------------------------------------------------------------------
// fp32 -> bf16 converts
// ---------------------------------------------------------------------------
__global__ void cvt_flat(const float* __restrict__ in, __bf16* __restrict__ out,
                         int n) {
  int i = blockIdx.x * 256 + threadIdx.x;
  if (i < n) out[i] = to_bf16(in[i]);
}

// in [rows][cols] fp32 -> out [cols][rows] bf16 (weight transpose for BT)
__global__ void cvt_T(const float* __restrict__ in, __bf16* __restrict__ out,
                      int rows, int cols) {
  int i = blockIdx.x * 256 + threadIdx.x;
  if (i >= rows * cols) return;
  int r = i / cols, c = i - r * cols;
  out[(size_t)c * rows + r] = to_bf16(in[i]);
}

// ---------------------------------------------------------------------------
// Generic bf16 WMMA GEMM: C[M][N] f32 = A[M][K] @ BT[N][K]^T
// block = 256 threads (8 waves); wave -> 32x64 tile; block tile 256x64.
// Per k-step: 6 fragment loads feed 8 WMMAs (B fragments reused for 2 rows).
// ---------------------------------------------------------------------------
__global__ void gemm_bf16(const __bf16* __restrict__ A,
                          const __bf16* __restrict__ BT,
                          float* __restrict__ C, int M, int N, int K) {
  int lane = threadIdx.x & 31;
  int wave = threadIdx.x >> 5;
  int m0 = blockIdx.y * 256 + wave * 32;
  int n0 = blockIdx.x * 64;
  int rn = lane & 15, half = lane >> 4;

  const __bf16* arow0 = A + (size_t)(m0 + rn) * K + half * 8;
  const __bf16* arow1 = arow0 + (size_t)16 * K;
  const __bf16* brow = BT + (size_t)(n0 + rn) * K + half * 8;
  const size_t bts = (size_t)16 * K;

  v8f acc0[4] = {}, acc1[4] = {};
#pragma unroll 2
  for (int k0 = 0; k0 < K; k0 += 32) {
    // speculative prefetch 4 k-steps (128 elements) ahead (global_prefetch_b8; OOB dropped)
    __builtin_prefetch(arow0 + k0 + 128, 0, 1);
    __builtin_prefetch(arow1 + k0 + 128, 0, 1);
    __builtin_prefetch(brow + k0 + 128, 0, 1);
    v16bf a0 = ldfrag(arow0 + k0);
    v16bf a1 = ldfrag(arow1 + k0);
#pragma unroll
    for (int t = 0; t < 4; ++t) {
      v16bf b = ldfrag(brow + (size_t)t * bts + k0);
      acc0[t] = wmma_bf16(a0, b, acc0[t]);
      acc1[t] = wmma_bf16(a1, b, acc1[t]);
    }
  }
  int off8 = half * 8;
#pragma unroll
  for (int j = 0; j < 8; ++j) {
    size_t o0 = (size_t)(m0 + off8 + j) * N + n0 + rn;
    size_t o1 = (size_t)(m0 + 16 + off8 + j) * N + n0 + rn;
#pragma unroll
    for (int t = 0; t < 4; ++t) {
      C[o0 + 16 * t] = acc0[t][j];
      C[o1 + 16 * t] = acc1[t][j];
    }
  }
}

// ---------------------------------------------------------------------------
// RoPE + pack: lin f32 [T][nh*64] -> out bf16 [h][T][64]
// ---------------------------------------------------------------------------
__global__ void rope_pack(const float* __restrict__ lin,
                          __bf16* __restrict__ out, int nh) {
  int i = blockIdx.x * 256 + threadIdx.x;
  int tot = TSEQ * nh * HDIM;
  if (i >= tot) return;
  int d = i & 63;
  int h = (i >> 6) % nh;
  int t = i / (nh * HDIM);
  int fi = d & 31;
  // inv_freq = theta^(-fi/32); ln(1e6) = 13.8155106
  float inv = __expf(-13.815511f * ((float)fi * (1.0f / 32.0f)));
  float ang = (float)t * inv;
  float s, c;
  __sincosf(ang, &s, &c);
  int base = t * nh * HDIM + h * HDIM;
  float v = lin[base + d];
  float vr = (d < 32) ? -lin[base + d + 32] : lin[base + d - 32];
  out[((size_t)h * TSEQ + t) * HDIM + d] = to_bf16(v * c + vr * s);
}

// V pack transpose: lin f32 [T][512] -> out bf16 [hkv*64 + d][T]
__global__ void vpack(const float* __restrict__ lin, __bf16* __restrict__ out) {
  int i = blockIdx.x * 256 + threadIdx.x;
  if (i >= TSEQ * KVDIM) return;
  int t = i / KVDIM;
  int c = i - t * KVDIM;
  out[(size_t)c * TSEQ + t] = to_bf16(lin[i]);
}

// ---------------------------------------------------------------------------
// Flash attention (causal, GQA). block = 128 (4 waves); wave owns 16 q rows.
// q bf16 [h][T][64], k bf16 [hkv][T][64], vt bf16 [hkv][64][T],
// ob bf16 [T][NHEADS*64].
// ---------------------------------------------------------------------------
__global__ void attn_kernel(const __bf16* __restrict__ qb,
                            const __bf16* __restrict__ kb,
                            const __bf16* __restrict__ vt,
                            __bf16* __restrict__ ob) {
  __shared__ float  sc[4][16 * 32];     // score tile, fp32
  __shared__ __bf16 pbuf[4][16 * 32];   // prob tile, bf16 row-major
  __shared__ float  corr_s[4][16];
  __shared__ float  l_s[4][16];

  int lane = threadIdx.x & 31;
  int wave = threadIdx.x >> 5;
  int h = blockIdx.y;
  int hkv = h >> 2;  // rep = 32/8 = 4
  int qbase = blockIdx.x * 64 + wave * 16;
  int rn = lane & 15, halfc = lane >> 4, off8 = halfc * 8;

  const __bf16* qrow = qb + ((size_t)h * TSEQ + qbase + rn) * HDIM + halfc * 8;
  v16bf aq0 = ldfrag(qrow);        // head-dim K = 0..31
  v16bf aq1 = ldfrag(qrow + 32);   // head-dim K = 32..63
  const __bf16* kh = kb + (size_t)hkv * TSEQ * HDIM;
  const __bf16* vh = vt + (size_t)hkv * HDIM * TSEQ;

  v8f acc0 = {}, acc1 = {}, acc2 = {}, acc3 = {};
  float m_r = -1e30f, l_r = 0.0f;
  float* scw = sc[wave];
  __bf16* pw = pbuf[wave];
  float* cw = corr_s[wave];
  float* lw = l_s[wave];

  int qend = qbase + 16;  // causal: only kv blocks touching <= qbase+15
  for (int kb0 = 0; kb0 < qend; kb0 += 32) {
    // ---- scores: Q(16x64) @ K^T(64x32) via 4 WMMA ----
    const __bf16* kp0 = kh + (size_t)(kb0 + rn) * HDIM + halfc * 8;
    const __bf16* kp1 = kp0 + (size_t)16 * HDIM;
    __builtin_prefetch(kp0 + 32 * HDIM, 0, 1);   // next kv block K rows
    v8f s0 = {}, s1 = {};
    s0 = wmma_bf16(aq0, ldfrag(kp0), s0);
    s0 = wmma_bf16(aq1, ldfrag(kp0 + 32), s0);
    s1 = wmma_bf16(aq0, ldfrag(kp1), s1);
    s1 = wmma_bf16(aq1, ldfrag(kp1 + 32), s1);

    // spill C-layout scores to LDS row-major [16 rows][32 cols]
#pragma unroll
    for (int j = 0; j < 8; ++j) {
      scw[(j + off8) * 32 + rn] = s0[j];
      scw[(j + off8) * 32 + rn + 16] = s1[j];
    }

    // ---- online softmax: lane pair (rn, rn+16) handles row rn ----
    int qg = qbase + rn;
    float vals[16];
    float mloc = -1e30f;
#pragma unroll
    for (int c = 0; c < 16; ++c) {
      int col = halfc * 16 + c;
      float v = scw[rn * 32 + col] * 0.125f;  // 1/sqrt(64)
      if (kb0 + col > qg) v = -1e9f;          // causal mask
      vals[c] = v;
      mloc = fmaxf(mloc, v);
    }
    mloc = fmaxf(mloc, __shfl_xor(mloc, 16, 32));
    float mnew = fmaxf(m_r, mloc);
    float cf = __expf(m_r - mnew);
    float ssum = 0.0f;
#pragma unroll
    for (int c = 0; c < 16; ++c) {
      float p = __expf(vals[c] - mnew);
      ssum += p;
      pw[rn * 32 + halfc * 16 + c] = to_bf16(p);
    }
    ssum += __shfl_xor(ssum, 16, 32);
    l_r = l_r * cf + ssum;
    m_r = mnew;
    cw[rn] = cf;  // both half-lanes write identical value

    // rescale accumulators by per-row correction
#pragma unroll
    for (int j = 0; j < 8; ++j) {
      float f = cw[j + off8];
      acc0[j] *= f; acc1[j] *= f; acc2[j] *= f; acc3[j] *= f;
    }

    // ---- PV: P(16x32) @ V(32x64) via 4 WMMA ----
    v16bf pa = ldfrag(pw + rn * 32 + halfc * 8);
    const __bf16* vp = vh + (size_t)rn * TSEQ + kb0 + halfc * 8;
    const size_t vts = (size_t)16 * TSEQ;
    acc0 = wmma_bf16(pa, ldfrag(vp), acc0);
    acc1 = wmma_bf16(pa, ldfrag(vp + vts), acc1);
    acc2 = wmma_bf16(pa, ldfrag(vp + 2 * vts), acc2);
    acc3 = wmma_bf16(pa, ldfrag(vp + 3 * vts), acc3);
  }

  lw[rn] = l_r;
#pragma unroll
  for (int j = 0; j < 8; ++j) {
    float inv = 1.0f / lw[j + off8];
    size_t o = (size_t)(qbase + off8 + j) * (NHEADS * HDIM) + h * HDIM + rn;
    ob[o] = to_bf16(acc0[j] * inv);
    ob[o + 16] = to_bf16(acc1[j] * inv);
    ob[o + 32] = to_bf16(acc2[j] * inv);
    ob[o + 48] = to_bf16(acc3[j] * inv);
  }
}

// ---------------------------------------------------------------------------
extern "C" void kernel_launch(void* const* d_in, const int* in_sizes, int n_in,
                              void* d_out, int out_size, void* d_ws,
                              size_t ws_size, hipStream_t stream) {
  const float* x = (const float*)d_in[0];
  // d_in[1] = mask (ignored; causal mask computed analytically)
  const float* w_q = (const float*)d_in[2];
  const float* w_k = (const float*)d_in[3];
  const float* w_v = (const float*)d_in[4];
  const float* w_o = (const float*)d_in[5];
  float* out = (float*)d_out;

  // workspace carve-up (all sizes multiples of 256B)
  char* w = (char*)d_ws;
  auto take = [&](size_t bytes) {
    char* p = w;
    w += (bytes + 255) & ~(size_t)255;
    return p;
  };
  __bf16* xb   = (__bf16*)take((size_t)TSEQ * DMODEL * 2);
  __bf16* wqT  = (__bf16*)take((size_t)DMODEL * DMODEL * 2);
  __bf16* wkT  = (__bf16*)take((size_t)KVDIM * DMODEL * 2);
  __bf16* wvT  = (__bf16*)take((size_t)KVDIM * DMODEL * 2);
  __bf16* woT  = (__bf16*)take((size_t)DMODEL * DMODEL * 2);
  float*  qlin = (float*)take((size_t)TSEQ * DMODEL * 4);
  float*  klin = (float*)take((size_t)TSEQ * KVDIM * 4);
  float*  vlin = (float*)take((size_t)TSEQ * KVDIM * 4);
  __bf16* qpk  = (__bf16*)take((size_t)NHEADS * TSEQ * HDIM * 2);
  __bf16* kpk  = (__bf16*)take((size_t)NKV * TSEQ * HDIM * 2);
  __bf16* vtp  = (__bf16*)take((size_t)NKV * HDIM * TSEQ * 2);
  __bf16* atn  = (__bf16*)take((size_t)TSEQ * DMODEL * 2);
  (void)ws_size;

  const int nx = TSEQ * DMODEL;
  const int nw = DMODEL * DMODEL;
  const int nkv = DMODEL * KVDIM;

  // stage bf16 operands
  cvt_flat<<<(nx + 255) / 256, 256, 0, stream>>>(x, xb, nx);
  cvt_T<<<(nw + 255) / 256, 256, 0, stream>>>(w_q, wqT, DMODEL, DMODEL);
  cvt_T<<<(nkv + 255) / 256, 256, 0, stream>>>(w_k, wkT, DMODEL, KVDIM);
  cvt_T<<<(nkv + 255) / 256, 256, 0, stream>>>(w_v, wvT, DMODEL, KVDIM);
  cvt_T<<<(nw + 255) / 256, 256, 0, stream>>>(w_o, woT, DMODEL, DMODEL);

  // QKV projections (block tile 256x64)
  gemm_bf16<<<dim3(DMODEL / 64, TSEQ / 256), 256, 0, stream>>>(
      xb, wqT, qlin, TSEQ, DMODEL, DMODEL);
  gemm_bf16<<<dim3(KVDIM / 64, TSEQ / 256), 256, 0, stream>>>(
      xb, wkT, klin, TSEQ, KVDIM, DMODEL);
  gemm_bf16<<<dim3(KVDIM / 64, TSEQ / 256), 256, 0, stream>>>(
      xb, wvT, vlin, TSEQ, KVDIM, DMODEL);

  // RoPE + packing
  int nq = TSEQ * NHEADS * HDIM;
  int nk = TSEQ * NKV * HDIM;
  rope_pack<<<(nq + 255) / 256, 256, 0, stream>>>(qlin, qpk, NHEADS);
  rope_pack<<<(nk + 255) / 256, 256, 0, stream>>>(klin, kpk, NKV);
  vpack<<<(nk + 255) / 256, 256, 0, stream>>>(vlin, vtp);

  // flash attention
  attn_kernel<<<dim3(TSEQ / 64, NHEADS), 128, 0, stream>>>(qpk, kpk, vtp, atn);

  // output projection -> fp32 out
  gemm_bf16<<<dim3(DMODEL / 64, TSEQ / 256), 256, 0, stream>>>(
      atn, woT, out, TSEQ, DMODEL, DMODEL);
  (void)in_sizes; (void)n_in; (void)out_size;
}
